// DINRecModel_67748814127822
// MI455X (gfx1250) — compile-verified, matrix operand order checked by reading, not run on hardware
//
#include <hip/hip_runtime.h>
#include <hip/hip_bf16.h>
#include <math.h>

// Model dims
#define Bsz  32
#define Mh   200
#define Nimp 50
#define Dd   128
#define Hh   64
#define ATTD 64
#define M1   256
#define M2   128

typedef __attribute__((ext_vector_type(16))) _Float16 v16h;
typedef __attribute__((ext_vector_type(8)))  _Float16 v8h;
typedef __attribute__((ext_vector_type(8)))  float    v8f;

__device__ __forceinline__ v8f wmma16x16x32(v16h a, v16h b, v8f c) {
    // (neg_a, A, neg_b, B, c_mod, C, reuse_a, reuse_b)
    return __builtin_amdgcn_wmma_f32_16x16x32_f16(false, a, false, b, (short)0, c,
                                                  false, false);
}

// Fragment load for f16 row-major data. Per the ISA 16-bit 16x32 layout, a
// lane's v16h operand is two contiguous 16-byte chunks:
//   halves 0..7  -> K = k0 + lg*8 + 0..7
//   halves 8..15 -> K = k0 + 16 + lg*8 + 0..7
// => two b128 loads instead of 16 scalar loads + converts.
__device__ __forceinline__ v16h load_frag_f16(const _Float16* __restrict__ rowp,
                                              int k0, int lg) {
    const _Float16* p = rowp + k0 + lg * 8;
    v8h lo = *(const v8h*)p;
    v8h hi = *(const v8h*)(p + 16);
    return __builtin_shufflevector(lo, hi, 0, 1, 2, 3, 4, 5, 6, 7,
                                   8, 9, 10, 11, 12, 13, 14, 15);
}

// ---------------------------------------------------------------------------
// f32 -> f16 conversion (coalesced, one-shot; bandwidth is trivial vs 23.3TB/s)
// ---------------------------------------------------------------------------
__global__ void __launch_bounds__(256) f32_to_f16_kernel(
    const float* __restrict__ in, _Float16* __restrict__ out, int n) {
    const int i = blockIdx.x * 256 + threadIdx.x;
    if (i < n) out[i] = (_Float16)in[i];
}

// ---------------------------------------------------------------------------
// C = A(MxK f16, row major) * W(CxK f16, row major)^T + bias.
// One wave per 16x64 output strip: A fragment loaded once per K-chunk and
// reused across 4 back-to-back WMMAs. Optional relu; optional dual f32/f16
// stores (f16 feeds the next GEMM directly).
// ---------------------------------------------------------------------------
__global__ void __launch_bounds__(32) gemm_wmma_f16(
    const _Float16* __restrict__ A, int lda,
    const _Float16* __restrict__ W, int ldw,
    const float* __restrict__ bias,
    float* __restrict__ C, _Float16* __restrict__ C16, int ldc,
    int K, int relu) {
    const int rt   = blockIdx.x;           // 16-row tile
    const int ct   = blockIdx.y;           // 64-col strip
    const int lane = threadIdx.x & 31;
    const int lg   = lane >> 4;
    const int ls   = lane & 15;

    const _Float16* Arow = A + (rt * 16 + ls) * lda;
    const _Float16* W0   = W + (ct * 64 + ls) * ldw;   // j advances by 16 rows

    v8f acc0 = {}, acc1 = {}, acc2 = {}, acc3 = {};
    for (int k0 = 0; k0 < K; k0 += 32) {
        __builtin_prefetch(Arow + k0 + 64, 0, 1);      // global_prefetch_b8
        const v16h a  = load_frag_f16(Arow, k0, lg);
        const v16h b0 = load_frag_f16(W0,              k0, lg);
        const v16h b1 = load_frag_f16(W0 + 16 * ldw,   k0, lg);
        const v16h b2 = load_frag_f16(W0 + 32 * ldw,   k0, lg);
        const v16h b3 = load_frag_f16(W0 + 48 * ldw,   k0, lg);
        acc0 = wmma16x16x32(a, b0, acc0);
        acc1 = wmma16x16x32(a, b1, acc1);
        acc2 = wmma16x16x32(a, b2, acc2);
        acc3 = wmma16x16x32(a, b3, acc3);
    }

    const int mb = lg * 8;
    v8f accs[4] = {acc0, acc1, acc2, acc3};
#pragma unroll
    for (int j = 0; j < 4; ++j) {
        const int   c0 = ct * 64 + j * 16;
        const float bv = bias ? bias[c0 + ls] : 0.0f;
#pragma unroll
        for (int r = 0; r < 8; ++r) {
            float v = accs[j][r] + bv;
            if (relu) v = fmaxf(v, 0.0f);
            const int off = (rt * 16 + mb + r) * ldc + c0 + ls;
            if (C)   C[off]   = v;
            if (C16) C16[off] = (_Float16)v;
        }
    }
}

// ---------------------------------------------------------------------------
// Bidirectional GRU recurrence. gx precomputed (f32). Hidden state resident in
// LDS for the whole 200-step scan: f32 copy for exact gate math, f16 copy for
// WMMA fragments (ds_load_b128). One workgroup (8 wave32) per direction.
// Per step: gh = h @ Whh^T + bhh (24 WMMA tiles), then gate math on VALU.
// Outputs go straight to cat16 (only a GEMM consumes them).
// ---------------------------------------------------------------------------
__global__ void __launch_bounds__(256) gru_bidir_kernel(
    const float* __restrict__ gx_f, const float* __restrict__ gx_b,
    const int* __restrict__ mask,
    const _Float16* __restrict__ Whh16_f, const float* __restrict__ bhh_f,
    const _Float16* __restrict__ Whh16_b, const float* __restrict__ bhh_b,
    _Float16* __restrict__ cat16) {
    __shared__ float    h_s[Bsz][Hh];        // 8 KB  (f32 state)
    __shared__ _Float16 h16[Bsz][Hh];        // 4 KB  (f16 mirror for WMMA)
    __shared__ float    gh_s[Bsz][3 * Hh];   // 24 KB

    const int       rev    = blockIdx.x;     // 0 = forward, 1 = backward
    const float*    gx     = rev ? gx_b : gx_f;
    const _Float16* Whh16  = rev ? Whh16_b : Whh16_f;
    const float*    bhh    = rev ? bhh_b : bhh_f;
    const int       dirOff = rev ? Hh : 0;

    for (int e = threadIdx.x; e < Bsz * Hh; e += 256) {
        h_s[e >> 6][e & 63] = 0.0f;
        h16[e >> 6][e & 63] = (_Float16)0.0f;
    }
    __syncthreads();

    const int wave = threadIdx.x >> 5;
    const int lane = threadIdx.x & 31;
    const int lg   = lane >> 4;
    const int ls   = lane & 15;

    for (int t = 0; t < Mh; ++t) {
        const int m = rev ? (Mh - 1 - t) : t;

        // gh = h @ Whh^T + bhh : 2 row tiles x 12 col tiles, K = 64
#pragma unroll
        for (int i = 0; i < 3; ++i) {
            const int tile = wave + i * 8;
            const int rt = tile / 12, ct = tile % 12;
            const _Float16* hrow = &h16[rt * 16 + ls][0];
            const _Float16* wrow = Whh16 + (ct * 16 + ls) * Hh;
            v8f acc = {};
#pragma unroll
            for (int kc = 0; kc < 2; ++kc) {
                const v16h a = load_frag_f16(hrow, kc * 32, lg);   // ds_load_b128
                const v16h b = load_frag_f16(wrow, kc * 32, lg);
                acc = wmma16x16x32(a, b, acc);
            }
            const int   col = ct * 16 + ls;
            const float bv  = bhh[col];
            const int   mb  = lg * 8;
#pragma unroll
            for (int r = 0; r < 8; ++r) gh_s[rt * 16 + mb + r][col] = acc[r] + bv;
        }
        __syncthreads();

        // gate math: 32*64 = 2048 elements / 256 threads
        for (int e = threadIdx.x; e < Bsz * Hh; e += 256) {
            const int b = e >> 6, j = e & 63;
            const float* g  = gx + (b * Mh + m) * (3 * Hh);
            const float  xr = g[j], xz = g[Hh + j], xn = g[2 * Hh + j];
            const float  hr = gh_s[b][j], hz = gh_s[b][Hh + j], hn = gh_s[b][2 * Hh + j];
            const float  hp = h_s[b][j];
            const float  r  = 1.0f / (1.0f + __expf(-(xr + hr)));
            const float  z  = 1.0f / (1.0f + __expf(-(xz + hz)));
            const float  nn = tanhf(xn + r * hn);
            const float  hc = (1.0f - z) * nn + z * hp;
            const bool   mk = mask[b * Mh + m] != 0;
            const float  hnew = mk ? hc : hp;
            h_s[b][j] = hnew;
            h16[b][j] = (_Float16)hnew;
            cat16[(b * Mh + m) * (2 * Hh) + dirOff + j] = (_Float16)(mk ? hc : 0.0f);
        }
        __syncthreads();
    }
}

// ---------------------------------------------------------------------------
// att[(b*N+n)*M + m] = mask ? relu(h_part[b,m]+c_part[b,n]) . Wa2 + ba2 : -inf
// ---------------------------------------------------------------------------
__global__ void __launch_bounds__(256) att_score_kernel(
    const float* __restrict__ h_part, const float* __restrict__ c_part,
    const float* __restrict__ Wa2, const float* __restrict__ ba2,
    const int* __restrict__ mask, float* __restrict__ att) {
    const int idx = blockIdx.x * 256 + threadIdx.x;
    if (idx >= Bsz * Mh * Nimp) return;
    const int n = idx % Nimp;
    const int m = (idx / Nimp) % Mh;
    const int b = idx / (Nimp * Mh);
    const float* hp = h_part + (b * Mh + m) * ATTD;
    const float* cp = c_part + (b * Nimp + n) * ATTD;
    float s = ba2[0];
#pragma unroll 8
    for (int a = 0; a < ATTD; ++a) s += fmaxf(hp[a] + cp[a], 0.0f) * Wa2[a];
    const bool mk = mask[b * Mh + m] != 0;
    att[(b * Nimp + n) * Mh + m] = mk ? s : -INFINITY;
}

// ---------------------------------------------------------------------------
// Softmax over M=200 per (b,n) row. One block of 128 threads per row.
// ---------------------------------------------------------------------------
__global__ void __launch_bounds__(128) softmax_kernel(float* __restrict__ att) {
    __shared__ float red[128];
    float* a = att + blockIdx.x * Mh;
    const int tid = threadIdx.x;

    float mx = -INFINITY;
    for (int m = tid; m < Mh; m += 128) mx = fmaxf(mx, a[m]);
    red[tid] = mx;
    __syncthreads();
    for (int s = 64; s > 0; s >>= 1) {
        if (tid < s) red[tid] = fmaxf(red[tid], red[tid + s]);
        __syncthreads();
    }
    mx = red[0];
    __syncthreads();

    float sum = 0.0f;
    for (int m = tid; m < Mh; m += 128) {
        const float e = __expf(a[m] - mx);
        a[m] = e;
        sum += e;
    }
    red[tid] = sum;
    __syncthreads();
    for (int s = 64; s > 0; s >>= 1) {
        if (tid < s) red[tid] += red[tid + s];
        __syncthreads();
    }
    const float inv = 1.0f / red[0];
    for (int m = tid; m < Mh; m += 128) a[m] *= inv;
}

// ---------------------------------------------------------------------------
// pooled = att_w^T @ hist_enc ; feat16 = concat(pooled, imps_emb) in f16
// (feat is only consumed by the Ws1 WMMA GEMM, so store f16 directly)
// ---------------------------------------------------------------------------
__global__ void __launch_bounds__(256) pooled_feat_kernel(
    const float* __restrict__ att, const float* __restrict__ hist_enc,
    const float* __restrict__ imps, _Float16* __restrict__ feat16) {
    const int idx = blockIdx.x * 256 + threadIdx.x;
    if (idx >= Bsz * Nimp * Dd) return;
    const int d   = idx & 127;
    const int row = idx >> 7;            // b*N + n
    const int b   = row / Nimp;
    const float* w  = att + row * Mh;
    const float* he = hist_enc + (b * Mh) * Dd + d;
    float s = 0.0f;
#pragma unroll 4
    for (int m = 0; m < Mh; ++m) s += w[m] * he[m * Dd];
    feat16[row * (2 * Dd) + d]      = (_Float16)s;
    feat16[row * (2 * Dd) + Dd + d] = (_Float16)imps[row * Dd + d];
}

// ---------------------------------------------------------------------------
// logits[row] = x2[row] . Ws3 + bs3
// ---------------------------------------------------------------------------
__global__ void __launch_bounds__(256) logits_kernel(
    const float* __restrict__ x2, const float* __restrict__ Ws3,
    const float* __restrict__ bs3, float* __restrict__ out) {
    const int row = blockIdx.x * 256 + threadIdx.x;
    if (row >= Bsz * Nimp) return;
    float s = bs3[0];
#pragma unroll 8
    for (int k = 0; k < M2; ++k) s += x2[row * M2 + k] * Ws3[k];
    out[row] = s;
}

// ---------------------------------------------------------------------------
static inline void conv16(const float* in, _Float16* out, int n, hipStream_t s) {
    f32_to_f16_kernel<<<(n + 255) / 256, 256, 0, s>>>(in, out, n);
}

extern "C" void kernel_launch(void* const* d_in, const int* in_sizes, int n_in,
                              void* d_out, int out_size, void* d_ws, size_t ws_size,
                              hipStream_t stream) {
    const float* user_hist = (const float*)d_in[0];
    const float* imps_emb  = (const float*)d_in[1];
    const int*   hist_mask = (const int*)d_in[2];
    const float* Wih_f  = (const float*)d_in[3];
    const float* Whh_f  = (const float*)d_in[4];
    const float* bih_f  = (const float*)d_in[5];
    const float* bhh_f  = (const float*)d_in[6];
    const float* Wih_b  = (const float*)d_in[7];
    const float* Whh_b  = (const float*)d_in[8];
    const float* bih_b  = (const float*)d_in[9];
    const float* bhh_b  = (const float*)d_in[10];
    const float* W_proj = (const float*)d_in[11];
    const float* b_proj = (const float*)d_in[12];
    const float* Wa1    = (const float*)d_in[13];
    const float* ba1    = (const float*)d_in[14];
    const float* Wa2    = (const float*)d_in[15];
    const float* ba2    = (const float*)d_in[16];
    const float* Ws1    = (const float*)d_in[17];
    const float* bs1    = (const float*)d_in[18];
    const float* Ws2    = (const float*)d_in[19];
    const float* bs2    = (const float*)d_in[20];
    const float* Ws3    = (const float*)d_in[21];
    const float* bs3    = (const float*)d_in[22];
    float* logits = (float*)d_out;

    // ---- workspace carve-up: f32 region then f16 region (16B aligned) ----
    float* pf = (float*)d_ws;
    float* gx_f     = pf; pf += (Bsz * Mh) * (3 * Hh);   // 6400 x 192
    float* gx_b     = pf; pf += (Bsz * Mh) * (3 * Hh);
    float* hist_enc = pf; pf += (Bsz * Mh) * Dd;         // 6400 x 128
    float* h_part   = pf; pf += (Bsz * Mh) * ATTD;       // 6400 x 64
    float* c_part   = pf; pf += (Bsz * Nimp) * ATTD;     // 1600 x 64
    float* att      = pf; pf += (Bsz * Nimp) * Mh;       // 1600 x 200
    float* x2       = pf; pf += (Bsz * Nimp) * M2;       // 1600 x 128

    _Float16* ph = (_Float16*)pf;
    _Float16* uh16    = ph; ph += (Bsz * Mh) * Dd;       // 6400 x 128
    _Float16* imps16  = ph; ph += (Bsz * Nimp) * Dd;     // 1600 x 128
    _Float16* Wihf16  = ph; ph += (3 * Hh) * Dd;         // 192 x 128
    _Float16* Wihb16  = ph; ph += (3 * Hh) * Dd;
    _Float16* Whhf16  = ph; ph += (3 * Hh) * Hh;         // 192 x 64
    _Float16* Whhb16  = ph; ph += (3 * Hh) * Hh;
    _Float16* Wproj16 = ph; ph += Dd * (2 * Hh);         // 128 x 128
    _Float16* Wa116   = ph; ph += ATTD * (2 * Dd);       // 64 x 256
    _Float16* Ws116   = ph; ph += M1 * (2 * Dd);         // 256 x 256
    _Float16* Ws216   = ph; ph += M2 * M1;               // 128 x 256
    _Float16* cat16   = ph; ph += (Bsz * Mh) * (2 * Hh); // 6400 x 128
    _Float16* he16    = ph; ph += (Bsz * Mh) * Dd;       // 6400 x 128
    _Float16* feat16  = ph; ph += (Bsz * Nimp) * (2 * Dd);
    _Float16* x116    = ph; ph += (Bsz * Nimp) * M1;

    // ---- 0) one-shot f32 -> f16 staging of activations & weights ----
    conv16(user_hist, uh16,   (Bsz * Mh) * Dd, stream);
    conv16(imps_emb,  imps16, (Bsz * Nimp) * Dd, stream);
    conv16(Wih_f,  Wihf16,  (3 * Hh) * Dd, stream);
    conv16(Wih_b,  Wihb16,  (3 * Hh) * Dd, stream);
    conv16(Whh_f,  Whhf16,  (3 * Hh) * Hh, stream);
    conv16(Whh_b,  Whhb16,  (3 * Hh) * Hh, stream);
    conv16(W_proj, Wproj16, Dd * (2 * Hh), stream);
    conv16(Wa1,    Wa116,   ATTD * (2 * Dd), stream);
    conv16(Ws1,    Ws116,   M1 * (2 * Dd), stream);
    conv16(Ws2,    Ws216,   M2 * M1, stream);

    // ---- 1) GRU input gate pre-projections: 6400x192, K=128 ----
    gemm_wmma_f16<<<dim3(400, 3), 32, 0, stream>>>(uh16, Dd, Wihf16, Dd, bih_f,
                                                   gx_f, nullptr, 3 * Hh, Dd, 0);
    gemm_wmma_f16<<<dim3(400, 3), 32, 0, stream>>>(uh16, Dd, Wihb16, Dd, bih_b,
                                                   gx_b, nullptr, 3 * Hh, Dd, 0);
    // ---- 2) bidirectional GRU scan (LDS-resident state, WMMA per step) ----
    gru_bidir_kernel<<<2, 256, 0, stream>>>(gx_f, gx_b, hist_mask,
                                            Whhf16, bhh_f, Whhb16, bhh_b, cat16);
    // ---- 3) hist_enc = cat @ W_proj^T + b_proj : 6400x128, K=128 ----
    gemm_wmma_f16<<<dim3(400, 2), 32, 0, stream>>>(cat16, 2 * Hh, Wproj16, 2 * Hh,
                                                   b_proj, hist_enc, he16, Dd,
                                                   2 * Hh, 0);
    // ---- 4) attention projections ----
    gemm_wmma_f16<<<dim3(400, 1), 32, 0, stream>>>(he16, Dd, Wa116, 2 * Dd,
                                                   (const float*)nullptr,
                                                   h_part, nullptr, ATTD, Dd, 0);
    gemm_wmma_f16<<<dim3(100, 1), 32, 0, stream>>>(imps16, Dd, Wa116 + Dd, 2 * Dd,
                                                   ba1, c_part, nullptr, ATTD, Dd, 0);
    // ---- 5) scores + masked softmax + pooling ----
    att_score_kernel<<<(Bsz * Mh * Nimp + 255) / 256, 256, 0, stream>>>(
        h_part, c_part, Wa2, ba2, hist_mask, att);
    softmax_kernel<<<Bsz * Nimp, 128, 0, stream>>>(att);
    pooled_feat_kernel<<<(Bsz * Nimp * Dd + 255) / 256, 256, 0, stream>>>(
        att, hist_enc, imps_emb, feat16);
    // ---- 6) MLP head (WMMA GEMMs with fused relu) ----
    gemm_wmma_f16<<<dim3(100, 4), 32, 0, stream>>>(feat16, 2 * Dd, Ws116, 2 * Dd,
                                                   bs1, nullptr, x116, M1, 2 * Dd, 1);
    gemm_wmma_f16<<<dim3(100, 2), 32, 0, stream>>>(x116, M1, Ws216, M1, bs2,
                                                   x2, nullptr, M2, M1, 1);
    logits_kernel<<<(Bsz * Nimp + 255) / 256, 256, 0, stream>>>(x2, Ws3, bs3, logits);
}
